// BlockInnerNet_59545426592023
// MI455X (gfx1250) — compile-verified, matrix operand order checked by reading, not run on hardware
//
#include <hip/hip_runtime.h>
#include <hip/hip_bf16.h>

// ---------------- problem constants (from reference) ----------------
constexpr int Bc  = 512;
constexpr int Tc  = 64;
constexpr int INc = 256;
constexpr int Hc  = 512;
constexpr int V1c = 101;
constexpr int Pc  = 8;
constexpr int CLSc = 512;
constexpr int REGc = 512;
constexpr int CCLD = 616;   // padded ld for [trans_prob | pf1] concat (613 -> 616)
constexpr int F2LD = 112;   // padded ld for feat2 (101 -> 112)

// ---------------- vector types (avoid HIP's ushortN names) ----------------
typedef __attribute__((ext_vector_type(16))) __bf16          v16bf;
typedef __attribute__((ext_vector_type(8)))  float           v8f;
typedef __attribute__((ext_vector_type(8)))  unsigned short  u16x8;
typedef __attribute__((ext_vector_type(4)))  unsigned short  u16x4;
typedef __attribute__((ext_vector_type(4)))  float           f32x4;

union BFrag {
  v16bf v;
  u16x8 h[2];
};

__device__ __forceinline__ unsigned short f2bf(float f) {
  unsigned int u = __float_as_uint(f);
  unsigned int r = u + 0x7FFFu + ((u >> 16) & 1u);   // round-to-nearest-even
  return (unsigned short)(r >> 16);
}

__device__ __forceinline__ float sigm(float x) {
  return 1.0f / (1.0f + __expf(-x));
}

// ---------------- f32 -> bf16 weight conversion ----------------
__global__ void cvt_f32_bf16(const float* __restrict__ src,
                             unsigned short* __restrict__ dst, int n) {
  int i = blockIdx.x * blockDim.x + threadIdx.x;
  for (; i < n; i += gridDim.x * blockDim.x) dst[i] = f2bf(src[i]);
}

__global__ void zero_f32(float* __restrict__ p, int n) {
  int i = blockIdx.x * blockDim.x + threadIdx.x;
  for (; i < n; i += gridDim.x * blockDim.x) p[i] = 0.0f;
}

// ---------------- WMMA GEMM: C = act(A1@W1^T + A2@W2^T + bias) ----------------
// A: M x K f32 row-major (M multiple of 32; here always 512)
// W: N x K bf16 row-major (ushort-encoded)
// wave tile: 16x32 (two 16x16 accumulators sharing one A fragment)
// workgroup (256 thr = 8 waves, 2 M-subtiles x 4 N-pairs) -> 32x128 tile
__global__ __launch_bounds__(256) void gemm_dual_bias_act(
    const float* __restrict__ A1, const unsigned short* __restrict__ W1, int K1,
    const float* __restrict__ A2, const unsigned short* __restrict__ W2, int K2,
    const float* __restrict__ bias, float* __restrict__ C,
    int N, int ldc, int act)
{
  __shared__ unsigned short As[32 * 40];    //  32x32 bf16 tile, padded stride 40
  __shared__ unsigned short Bs[128 * 40];   // 128x32 bf16 tile

  const int tid  = threadIdx.x;
  const int lane = tid & 31;
  const int wave = tid >> 5;
  const int wm  = wave & 1;        // 0..1 : M sub-tile (16 rows each)
  const int wnp = wave >> 1;       // 0..3 : N pair sub-tile (32 cols each)
  const int m0 = blockIdx.y * 32;
  const int n0 = blockIdx.x * 128;

  v8f acc0, acc1;
#pragma unroll
  for (int i = 0; i < 8; ++i) { acc0[i] = 0.0f; acc1[i] = 0.0f; }

  for (int pass = 0; pass < 2; ++pass) {
    const float* A = pass ? A2 : A1;
    const unsigned short* W = pass ? W2 : W1;
    const int K = pass ? K2 : K1;
    if (A == nullptr) continue;

    for (int k0 = 0; k0 < K; k0 += 32) {
      // stage A tile (f32 -> bf16): 4 elements / thread (32x32 = 1024)
      {
        int flat = tid * 4, r = flat >> 5, c = flat & 31;
        f32x4 av = *(const f32x4*)(A + (size_t)(m0 + r) * K + k0 + c);
        u16x4 bv;
        bv[0] = f2bf(av[0]); bv[1] = f2bf(av[1]);
        bv[2] = f2bf(av[2]); bv[3] = f2bf(av[3]);
        *(u16x4*)&As[r * 40 + c] = bv;
      }
      // stage W tile (already bf16): 16 elements / thread (128x32 = 4096)
      {
        int flat = tid * 16, r = flat >> 5, c = flat & 31;  // c in {0,16}
        int n = n0 + r;
        u16x8 wv0 = {0, 0, 0, 0, 0, 0, 0, 0};
        u16x8 wv1 = {0, 0, 0, 0, 0, 0, 0, 0};
        if (n < N) {
          const unsigned short* wp = W + (size_t)n * K + k0 + c;
          wv0 = *(const u16x8*)(wp);
          wv1 = *(const u16x8*)(wp + 8);
        }
        *(u16x8*)&Bs[r * 40 + c]     = wv0;
        *(u16x8*)&Bs[r * 40 + c + 8] = wv1;
      }
      __syncthreads();

      // fragment loads per ISA 16-bit A/B layout:
      // lane L: row = L%16 ; K chunks [kh, kh+8) and [16+kh, ...)
      BFrag a, b0, b1;
      const int rA  = wm * 16 + (lane & 15);
      const int rB0 = wnp * 32 + (lane & 15);
      const int rB1 = rB0 + 16;
      const int kh  = (lane >> 4) << 3;
      a.h[0]  = *(const u16x8*)&As[rA * 40 + kh];
      a.h[1]  = *(const u16x8*)&As[rA * 40 + 16 + kh];
      b0.h[0] = *(const u16x8*)&Bs[rB0 * 40 + kh];
      b0.h[1] = *(const u16x8*)&Bs[rB0 * 40 + 16 + kh];
      b1.h[0] = *(const u16x8*)&Bs[rB1 * 40 + kh];
      b1.h[1] = *(const u16x8*)&Bs[rB1 * 40 + 16 + kh];

      acc0 = __builtin_amdgcn_wmma_f32_16x16x32_bf16(
          false, a.v, false, b0.v, (short)0, acc0, false, false);
      acc1 = __builtin_amdgcn_wmma_f32_16x16x32_bf16(
          false, a.v, false, b1.v, (short)0, acc1, false, false);
      __syncthreads();
    }
  }

  // C layout: VGPR r, lane L -> row = r + 8*(L/16), col = L%16
  const int mBase = m0 + wm * 16 + ((lane >> 4) << 3);
#pragma unroll
  for (int i = 0; i < 2; ++i) {
    const int nIdx = n0 + wnp * 32 + i * 16 + (lane & 15);
    const v8f& acc = i ? acc1 : acc0;
    if (nIdx < N) {
      float bb = bias ? bias[nIdx] : 0.0f;
#pragma unroll
      for (int r = 0; r < 8; ++r) {
        float v = acc[r] + bb;
        if (act) v = fmaxf(v, 0.0f);
        C[(size_t)(mBase + r) * ldc + nIdx] = v;
      }
    }
  }
}

// ---------------- LSTM pointwise cell update ----------------
__global__ void lstm_update(const float* __restrict__ gates,
                            float* __restrict__ h, float* __restrict__ c) {
  int i = blockIdx.x * blockDim.x + threadIdx.x;
  if (i >= Bc * Hc) return;
  int b = i / Hc, j = i % Hc;
  const float* g = gates + (size_t)b * 4 * Hc;
  float ig = sigm(g[j]);
  float fg = sigm(g[Hc + j]);
  float gg = tanhf(g[2 * Hc + j]);
  float og = sigm(g[3 * Hc + j]);
  float cn = fg * c[i] + ig * gg;
  c[i] = cn;
  h[i] = og * tanhf(cn);
}

// ---------------- embedding gather: xcur = embed[y[:, t-1]] ----------------
__global__ void gather_embed(const float* __restrict__ embed,
                             const int* __restrict__ y, int t,
                             float* __restrict__ xcur) {
  int i = blockIdx.x * blockDim.x + threadIdx.x;
  if (i >= Bc * INc) return;
  int b = i / INc, col = i % INc;
  int tok = y[b * Tc + (t - 1)];
  xcur[i] = embed[(size_t)tok * INc + col];
}

// -------- softmax head: log_softmax -> out1, softmax -> cc[:, 0:101] --------
__global__ void softmax_head(const float* __restrict__ feat2,
                             float* __restrict__ out1,
                             float* __restrict__ cc, int t) {
  int b = blockIdx.x, tid = threadIdx.x;   // 128 threads
  __shared__ float red[128];
  float v = (tid < V1c) ? feat2[(size_t)b * F2LD + tid] : -1e30f;
  red[tid] = v;
  __syncthreads();
  for (int s = 64; s > 0; s >>= 1) {
    if (tid < s) red[tid] = fmaxf(red[tid], red[tid + s]);
    __syncthreads();
  }
  float mx = red[0];
  __syncthreads();
  float e = (tid < V1c) ? __expf(v - mx) : 0.0f;
  red[tid] = e;
  __syncthreads();
  for (int s = 64; s > 0; s >>= 1) {
    if (tid < s) red[tid] += red[tid + s];
    __syncthreads();
  }
  float sum = red[0];
  if (tid < V1c) {
    out1[((size_t)b * Tc + t) * V1c + tid] = (v - mx) - __logf(sum);
    cc[(size_t)b * CCLD + tid] = e / sum;
  }
}

// -------- gathered r2 head: only the P=8 rows selected by y[b,t] --------
__global__ void param_head(const float* __restrict__ cc,
                           const float* __restrict__ r2w,
                           const float* __restrict__ r2b,
                           const int* __restrict__ y,
                           float* __restrict__ out2, int t) {
  int b = blockIdx.x, tid = threadIdx.x;   // 128 = 8 groups x 16 lanes
  int p = tid >> 4, lane = tid & 15;
  int idx = y[b * Tc + t];
  const int Kc = V1c + REGc;               // 613
  const float* w = r2w + (size_t)(idx * Pc + p) * Kc;
  const float* c = cc + (size_t)b * CCLD;
  float s = 0.0f;
  for (int j = lane; j < Kc; j += 16) s += c[j] * w[j];
  __shared__ float red[128];
  red[tid] = s;
  __syncthreads();
  for (int st = 8; st > 0; st >>= 1) {
    if (lane < st) red[tid] += red[tid + st];
    __syncthreads();
  }
  if (lane == 0) out2[((size_t)b * Tc + t) * Pc + p] = red[tid] + r2b[idx * Pc + p];
}

// ---------------- host driver ----------------
extern "C" void kernel_launch(void* const* d_in, const int* in_sizes, int n_in,
                              void* d_out, int out_size, void* d_ws, size_t ws_size,
                              hipStream_t stream) {
  const float* x     = (const float*)d_in[0];
  const int*   y     = (const int*)  d_in[1];
  const float* embed = (const float*)d_in[2];
  const float* wih0  = (const float*)d_in[3];
  const float* whh0  = (const float*)d_in[4];
  const float* wih1  = (const float*)d_in[5];
  const float* whh1  = (const float*)d_in[6];
  const float* l1w   = (const float*)d_in[7];
  const float* l1b   = (const float*)d_in[8];
  const float* l2w   = (const float*)d_in[9];
  const float* l2b   = (const float*)d_in[10];
  const float* r1w   = (const float*)d_in[11];
  const float* r1b   = (const float*)d_in[12];
  const float* r2w   = (const float*)d_in[13];
  const float* r2b   = (const float*)d_in[14];

  float* out1 = (float*)d_out;                       // (B,T,V1)
  float* out2 = out1 + (size_t)Bc * Tc * V1c;        // (B,T,P)

  // workspace carve-up
  char* wp = (char*)d_ws;
  auto alloc = [&](size_t bytes) -> char* {
    char* r = wp;
    wp += (bytes + 255) & ~(size_t)255;
    return r;
  };
  unsigned short* wih0_bf = (unsigned short*)alloc((size_t)4 * Hc * INc * 2);
  unsigned short* whh0_bf = (unsigned short*)alloc((size_t)4 * Hc * Hc * 2);
  unsigned short* wih1_bf = (unsigned short*)alloc((size_t)4 * Hc * Hc * 2);
  unsigned short* whh1_bf = (unsigned short*)alloc((size_t)4 * Hc * Hc * 2);
  unsigned short* l1w_bf  = (unsigned short*)alloc((size_t)CLSc * Hc * 2);
  unsigned short* l2w_bf  = (unsigned short*)alloc((size_t)V1c * CLSc * 2);
  unsigned short* r1w_bf  = (unsigned short*)alloc((size_t)REGc * Hc * 2);
  float* states = (float*)alloc((size_t)4 * Bc * Hc * 4);  // h0,c0,h1,c1
  float* h0 = states;
  float* c0 = h0 + (size_t)Bc * Hc;
  float* h1 = c0 + (size_t)Bc * Hc;
  float* c1 = h1 + (size_t)Bc * Hc;
  float* gates  = (float*)alloc((size_t)Bc * 4 * Hc * 4);
  float* xcur   = (float*)alloc((size_t)Bc * INc * 4);
  float* hidden = (float*)alloc((size_t)Bc * CLSc * 4);
  float* feat2  = (float*)alloc((size_t)Bc * F2LD * 4);
  float* cc     = (float*)alloc((size_t)Bc * CCLD * 4);

  // weights -> bf16 once per call
  auto cvt = [&](const float* s, unsigned short* d, int n) {
    cvt_f32_bf16<<<(n + 255) / 256, 256, 0, stream>>>(s, d, n);
  };
  cvt(wih0, wih0_bf, 4 * Hc * INc);
  cvt(whh0, whh0_bf, 4 * Hc * Hc);
  cvt(wih1, wih1_bf, 4 * Hc * Hc);
  cvt(whh1, whh1_bf, 4 * Hc * Hc);
  cvt(l1w,  l1w_bf,  CLSc * Hc);
  cvt(l2w,  l2w_bf,  V1c * CLSc);
  cvt(r1w,  r1w_bf,  REGc * Hc);

  // zero h/c every call (workspace is not re-poisoned between replays)
  {
    int n = 4 * Bc * Hc;
    zero_f32<<<(n + 255) / 256, 256, 0, stream>>>(states, n);
  }

  const dim3 blk(256);
  const dim3 gGates(4 * Hc / 128, Bc / 32);          // N=2048 -> 16 x 16
  const dim3 gH512(Hc / 128, Bc / 32);               // N=512  -> 4 x 16
  const dim3 gF2((V1c + 127) / 128, Bc / 32);        // N=101  -> 1 x 16
  const int nBH = Bc * Hc;

  for (int t = 0; t < Tc; ++t) {
    const float* a0 = x;
    if (t > 0) {
      gather_embed<<<(Bc * INc + 255) / 256, 256, 0, stream>>>(embed, y, t, xcur);
      a0 = xcur;
    }
    // layer 0 gates + update
    gemm_dual_bias_act<<<gGates, blk, 0, stream>>>(
        a0, wih0_bf, INc, h0, whh0_bf, Hc, nullptr, gates, 4 * Hc, 4 * Hc, 0);
    lstm_update<<<(nBH + 255) / 256, 256, 0, stream>>>(gates, h0, c0);
    // layer 1 gates + update
    gemm_dual_bias_act<<<gGates, blk, 0, stream>>>(
        h0, wih1_bf, Hc, h1, whh1_bf, Hc, nullptr, gates, 4 * Hc, 4 * Hc, 0);
    lstm_update<<<(nBH + 255) / 256, 256, 0, stream>>>(gates, h1, c1);

    // heads on h1 (== outs[t])
    gemm_dual_bias_act<<<gH512, blk, 0, stream>>>(
        h1, l1w_bf, Hc, nullptr, nullptr, 0, l1b, hidden, CLSc, CLSc, 1);
    gemm_dual_bias_act<<<gF2, blk, 0, stream>>>(
        hidden, l2w_bf, CLSc, nullptr, nullptr, 0, l2b, feat2, V1c, F2LD, 0);
    softmax_head<<<Bc, 128, 0, stream>>>(feat2, out1, cc, t);
    gemm_dual_bias_act<<<gH512, blk, 0, stream>>>(
        h1, r1w_bf, Hc, nullptr, nullptr, 0, r1b, cc + V1c, REGc, CCLD, 1);
    param_head<<<Bc, 128, 0, stream>>>(cc, r2w, r2b, y, out2, t);
  }
}
